// cnn_transformer_lstm_11433202942310
// MI455X (gfx1250) — matrix operand order repaired and checked by reading, not compiled, not run on hardware
//
#include <hip/hip_runtime.h>
#include <hip/hip_bf16.h>

// ---------------- problem constants ----------------
#define BB   8
#define NSZv 4
#define TT   120
#define CCH  19
#define LLEN 200
#define DDIM 32
#define NHEAD 8
#define HDIM 4
#define DFFN 128
#define HIDN 16
#define BT   (BB * NSZv * TT)      // 3840 token batches
#define P1LD 22                    // padded conv1-activation row (20 used + 2 zero)
#define P1SZ (DDIM * P1LD)         // per-item p1 tile (halves)

typedef _Float16 half_t;
typedef half_t v16h __attribute__((ext_vector_type(16)));
typedef float  v8f  __attribute__((ext_vector_type(8)));

// Allowed-attention bitmasks: bit kj of ALLOW[qi] == 1 -> (qi,kj) allowed.
__constant__ unsigned int ALLOW[20] = {
    0x0001Fu, 0x00073u, 0x001DDu, 0x0033Du, 0x0023Bu,
    0x0067Au, 0x00C76u, 0x03984u, 0x0779Cu, 0x0E738u,
    0x1CF70u, 0x18CC0u, 0x33180u, 0x2F380u, 0x6E700u,
    0x5EE00u, 0x59C00u, 0x67000u, 0x7C000u, 0xFFFFFu
};

// ---------------- WMMA fragment helpers (wave32, 16x16x32 f16->f32) --------
// A 16x32 (MxK): lanes 0-15 hold rows M=lane, K=0..15 (v[0..15]);
// lanes 16-31 hold same rows with K=16..31.
__device__ __forceinline__ v16h ldA_f32(const float* src, int ld, int m0, int k0, int lane) {
    int r  = m0 + (lane & 15);
    int kb = k0 + ((lane >> 4) << 4);
    v16h a;
#pragma unroll
    for (int i = 0; i < 16; ++i) a[i] = (half_t)src[r * ld + kb + i];
    return a;
}

__device__ __forceinline__ v16h ldA_f16(const half_t* src, int ld, int m0, int k0, int lane) {
    int r  = m0 + (lane & 15);
    int kb = k0 + ((lane >> 4) << 4);
    v16h a;
#pragma unroll
    for (int i = 0; i < 16; ++i) a[i] = src[r * ld + kb + i];
    return a;
}

// B 32x16 (KxN) built from a row-major weight W[n][k] (y = x @ W^T):
// lane n = lane&15 holds column n0+n; lanes 0-15 carry K=k0..k0+15,
// lanes 16-31 carry K=k0+16..k0+31.
__device__ __forceinline__ v16h ldB_wT(const half_t* W, int ld, int n0, int k0, int lane) {
    int n  = n0 + (lane & 15);
    int kb = k0 + ((lane >> 4) << 4);
    v16h b;
#pragma unroll
    for (int i = 0; i < 16; ++i) b[i] = W[n * ld + kb + i];
    return b;
}

// B 32x16 gather for conv2-as-GEMM with tap dim padded to 8:
// column n encodes (item = n>>3, pos = n&7); row k encodes
// (ic = k>>3, kk = k&7); element = p1[item][ic][2*pos + kk].
// k0 is a compile-time-unrolled multiple of 32 and kb is 16-aligned, so all
// shifts/masks constant-fold (no integer division in the address path).
__device__ __forceinline__ v16h ldB_conv8(const half_t* p1, int k0, int lane) {
    const int n    = lane & 15;
    const int item = n >> 3;
    const int t    = n & 7;
    const half_t* base = p1 + item * P1SZ + 2 * t;
    const int kb = k0 + ((lane >> 4) << 4);
    v16h b;
#pragma unroll
    for (int i = 0; i < 16; ++i) {
        const int k = kb + i;
        b[i] = base[(k >> 3) * P1LD + (k & 7)];
    }
    return b;
}

// C/D 16x16 f32: element j of v8f is (M = m0 + (lane>>4)*8 + j, N = n0 + lane&15)
__device__ __forceinline__ void stC_f16(half_t* dst, int ld, int m0, int n0, int lane, v8f c) {
    int n  = n0 + (lane & 15);
    int mb = m0 + ((lane >> 4) << 3);
#pragma unroll
    for (int j = 0; j < 8; ++j) dst[(mb + j) * ld + n] = (half_t)c[j];
}

__device__ __forceinline__ v8f wmma16x16x32(v16h a, v16h b, v8f c) {
    return __builtin_amdgcn_wmma_f32_16x16x32_f16(false, a, false, b, (short)0, c, false, false);
}

__device__ __forceinline__ float sigm(float x) { return 1.f / (1.f + expf(-x)); }

// =====================================================================
// Kernel 1: per-channel CNN.  One wave per pair of (sample, channel)
// items; lane = output channel.  conv1 is VALU (K=5 taps, cheap);
// conv2 (the dominant 6-GFLOP GEMM: M=32, K=32ch x 8 padded taps,
// N=2 items x 8 pos) runs as 2x8 = 16 v_wmma_f32_16x16x32_f16 per pair,
// with pad taps carried as zero weights so the B gather needs no divides.
// =====================================================================
__global__ __launch_bounds__(256) void k_cnn(
    const float* __restrict__ x, const float* __restrict__ pos,
    const float* __restrict__ c1w, const float* __restrict__ c1b,
    const float* __restrict__ c2w, const float* __restrict__ c2b,
    float* __restrict__ tokbuf)
{
    // s_in (pre-conv2) and s_o2 (post-conv2) overlap; phases separated by barriers.
    __shared__ __align__(16) char s_scratch[16 * LLEN * 4];     // 12.8KB -> fits s_in only; the s_o2 view spans 16KB
    __shared__ half_t s_p1[16][P1SZ];                           // conv1 activations, f16, padded rows
    __shared__ half_t s_w2h[DDIM * DDIM * 8];                   // conv2 weights, f16, taps padded 5->8

    float (*s_in)[LLEN]     = reinterpret_cast<float(*)[LLEN]>(s_scratch);     // [16][200]
    float (*s_o2)[DDIM][16] = reinterpret_cast<float(*)[DDIM][16]>(s_scratch); // [8][32][16]

    const int tid  = threadIdx.x;
    const int w    = tid >> 5;
    const int lane = tid & 31;

    // stage conv2 weights: s_w2h[oc][ic*8+kk] = (kk<5) ? c2w[oc][ic][kk] : 0
    for (int i = tid; i < DDIM * DDIM * 8; i += 256) {
        const int kk = i & 7;
        const int ic = (i >> 3) & 31;
        const int oc = i >> 8;
        s_w2h[i] = (kk < 5) ? (half_t)c2w[(oc * DDIM + ic) * 5 + kk] : (half_t)0.f;
    }

    const int item0 = blockIdx.x * 16 + w * 2;   // exact: grid*16 == BT*CCH
#pragma unroll
    for (int e = 0; e < 2; ++e) {
        const int item = item0 + e;
        const int samp = item / CCH;
        const int ch   = item - samp * CCH;
        const float* xp = x + (size_t)samp * CCH * LLEN + (size_t)ch * LLEN;
        for (int i = lane; i < LLEN; i += 32) s_in[w * 2 + e][i] = xp[i];
    }
    __syncthreads();

    // conv1 (k=5, stride 5) + relu + maxpool2 : 40 -> 20 values per channel
    float wk1[5];
#pragma unroll
    for (int k = 0; k < 5; ++k) wk1[k] = c1w[lane * 5 + k];
    const float b1v = c1b[lane];
#pragma unroll
    for (int e = 0; e < 2; ++e) {
        const float* si = s_in[w * 2 + e];
        half_t* prow = &s_p1[w * 2 + e][lane * P1LD];
#pragma unroll
        for (int t = 0; t < 20; ++t) {
            float a0 = b1v, a1 = b1v;
            const int base = t * 10;
#pragma unroll
            for (int k = 0; k < 5; ++k) {
                a0 += wk1[k] * si[base + k];
                a1 += wk1[k] * si[base + 5 + k];
            }
            a0 = fmaxf(a0, 0.f); a1 = fmaxf(a1, 0.f);
            prow[t] = (half_t)fmaxf(a0, a1);
        }
        prow[20] = (half_t)0.f;    // zero the pad tail (read by padded-tap gather)
        prow[21] = (half_t)0.f;
    }
    __syncthreads();

    // conv2 as WMMA GEMM: out[oc][n] = sum_k W2pad[oc][k] * B[k][n], K=256
    v8f acc0 = {}, acc1 = {};
    const half_t* p1b = &s_p1[w * 2][0];
#pragma unroll
    for (int kc = 0; kc < 8; ++kc) {
        v16h b  = ldB_conv8(p1b, kc * 32, lane);
        v16h a0 = ldA_f16(s_w2h, 256, 0,  kc * 32, lane);
        v16h a1 = ldA_f16(s_w2h, 256, 16, kc * 32, lane);
        acc0 = wmma16x16x32(a0, b, acc0);
        acc1 = wmma16x16x32(a1, b, acc1);
    }
    __syncthreads();   // all waves done reading s_in overlay before o2 writes

    // bias + relu, scatter to LDS for the cross-lane pooling step
    {
        const int n  = lane & 15;
        const int hb = (lane >> 4) << 3;
#pragma unroll
        for (int j = 0; j < 8; ++j) {
            const float v0 = acc0[j] + c2b[hb + j];
            s_o2[w][hb + j][n] = fmaxf(v0, 0.f);
            const float v1 = acc1[j] + c2b[16 + hb + j];
            s_o2[w][16 + hb + j][n] = fmaxf(v1, 0.f);
        }
    }
    __syncthreads();

    // maxpool2 (8 -> 4) + mean, then write feat + pos_emb
#pragma unroll
    for (int e = 0; e < 2; ++e) {
        float m = 0.f;
#pragma unroll
        for (int t = 0; t < 4; ++t)
            m += fmaxf(s_o2[w][lane][e * 8 + 2 * t], s_o2[w][lane][e * 8 + 2 * t + 1]);
        m *= 0.25f;
        const int item = item0 + e;
        const int samp = item / CCH;
        const int ch   = item - samp * CCH;
        tokbuf[(size_t)samp * CCH * DDIM + ch * DDIM + lane] = m + pos[ch * DDIM + lane];
    }
}

// =====================================================================
// Kernel 2: transformer encoder layer.  One wave32 per sample,
// 2 waves / 64-thread block.  All GEMMs via v_wmma_f32_16x16x32_f16.
// =====================================================================
__global__ __launch_bounds__(64) void k_xform(
    const float* __restrict__ tokbuf, const float* __restrict__ pos,
    const float* __restrict__ wq, const float* __restrict__ bq,
    const float* __restrict__ wk, const float* __restrict__ bk,
    const float* __restrict__ wv, const float* __restrict__ bv,
    const float* __restrict__ wo, const float* __restrict__ bo,
    const float* __restrict__ ln1g, const float* __restrict__ ln1b,
    const float* __restrict__ w1, const float* __restrict__ b1,
    const float* __restrict__ w2, const float* __restrict__ b2,
    const float* __restrict__ ln2g, const float* __restrict__ ln2b,
    float* __restrict__ hc_out, float* __restrict__ hm_out)
{
    // block-shared f16 weights
    __shared__ half_t s_wq[DDIM * DDIM], s_wk[DDIM * DDIM];
    __shared__ half_t s_wv[DDIM * DDIM], s_wo[DDIM * DDIM];
    __shared__ half_t s_w1[DFFN * DDIM], s_w2[DDIM * DFFN];
    // per-sample slots (2 waves)
    __shared__ float  s_tok[2][32 * DDIM];
    __shared__ half_t s_q[2][32 * DDIM], s_k[2][32 * DDIM], s_v[2][32 * DDIM];
    __shared__ half_t s_ao[2][32 * DDIM], s_hs[2][32 * DDIM];

    const int tid  = threadIdx.x;
    const int wid  = tid >> 5;
    const int lane = tid & 31;

    for (int i = tid; i < DDIM * DDIM; i += 64) {
        s_wq[i] = (half_t)wq[i]; s_wk[i] = (half_t)wk[i];
        s_wv[i] = (half_t)wv[i]; s_wo[i] = (half_t)wo[i];
    }
    for (int i = tid; i < DFFN * DDIM; i += 64) {
        s_w1[i] = (half_t)w1[i];
        s_w2[i] = (half_t)w2[i];
    }

    const int s = blockIdx.x * 2 + wid;             // sample id < BT
    float*  tok = s_tok[wid];
    half_t* sq  = s_q[wid];
    half_t* sk  = s_k[wid];
    half_t* sv  = s_v[wid];
    half_t* sao = s_ao[wid];
    half_t* shs = s_hs[wid];

    // ---- build padded 32x32 token tile (rows 20..31 zero) ----
    const float* tb = tokbuf + (size_t)s * CCH * DDIM;
    for (int r = 0; r < CCH; ++r) tok[r * DDIM + lane] = tb[r * DDIM + lane];
    tok[19 * DDIM + lane] = pos[19 * DDIM + lane];
    for (int r = 20; r < 32; ++r) tok[r * DDIM + lane] = 0.f;
    for (int r = 0; r < 32; ++r)  sao[r * DDIM + lane] = (half_t)0.f;
    __syncthreads();

    // ---- Q, K, V projections (12 WMMAs) ----
    {
        v16h aT0 = ldA_f32(tok, DDIM, 0, 0, lane);
        v16h aT1 = ldA_f32(tok, DDIM, 16, 0, lane);
        const half_t* Ws[3]   = { s_wq, s_wk, s_wv };
        const float*  Bs[3]   = { bq, bk, bv };
        half_t*       Ds[3]   = { sq, sk, sv };
#pragma unroll
        for (int p = 0; p < 3; ++p) {
#pragma unroll
            for (int mt = 0; mt < 2; ++mt) {
                v16h a = (mt == 0) ? aT0 : aT1;
#pragma unroll
                for (int nt = 0; nt < 2; ++nt) {
                    v16h b = ldB_wT(Ws[p], DDIM, nt * 16, 0, lane);
                    v8f  c = {};
                    c = wmma16x16x32(a, b, c);
                    const float bb = Bs[p][nt * 16 + (lane & 15)];
#pragma unroll
                    for (int j = 0; j < 8; ++j) c[j] += bb;
                    stC_f16(Ds[p], DDIM, mt * 16, nt * 16, lane, c);
                }
            }
        }
    }
    __syncthreads();

    // ---- masked multi-head attention (VALU; HD=4 too small for WMMA) ----
#pragma unroll
    for (int it = 0; it < 5; ++it) {                 // 8 heads * 20 q-rows / 32 lanes
        const int idx = it * 32 + lane;
        const int h   = idx / 20;
        const int qi  = idx - h * 20;
        const half_t* qp = sq + qi * DDIM + h * HDIM;
        const unsigned int am = ALLOW[qi];
        float sc[20];
        float mx = -1e30f;
#pragma unroll
        for (int kj = 0; kj < 20; ++kj) {
            float dot = 0.f;
#pragma unroll
            for (int d = 0; d < HDIM; ++d)
                dot += (float)qp[d] * (float)sk[kj * DDIM + h * HDIM + d];
            dot *= 0.5f;                             // 1/sqrt(HD)
            const float vsc = ((am >> kj) & 1u) ? dot : -1e9f;
            sc[kj] = vsc;
            mx = fmaxf(mx, vsc);
        }
        float ssum = 0.f;
#pragma unroll
        for (int kj = 0; kj < 20; ++kj) { const float e = expf(sc[kj] - mx); sc[kj] = e; ssum += e; }
        const float inv = 1.f / ssum;
#pragma unroll
        for (int d = 0; d < HDIM; ++d) {
            float o = 0.f;
#pragma unroll
            for (int kj = 0; kj < 20; ++kj)
                o += sc[kj] * (float)sv[kj * DDIM + h * HDIM + d];
            sao[qi * DDIM + h * HDIM + d] = (half_t)(o * inv);
        }
    }
    __syncthreads();

    // ---- output projection + residual (4 WMMAs) ----
#pragma unroll
    for (int mt = 0; mt < 2; ++mt) {
        v16h a = ldA_f16(sao, DDIM, mt * 16, 0, lane);
#pragma unroll
        for (int nt = 0; nt < 2; ++nt) {
            v16h b = ldB_wT(s_wo, DDIM, nt * 16, 0, lane);
            v8f  c = {};
            c = wmma16x16x32(a, b, c);
            const int   n  = nt * 16 + (lane & 15);
            const int   mb = mt * 16 + ((lane >> 4) << 3);
            const float bb = bo[n];
#pragma unroll
            for (int j = 0; j < 8; ++j) tok[(mb + j) * DDIM + n] += c[j] + bb;
        }
    }
    __syncthreads();

    // ---- LayerNorm 1 (wave32 shuffle reductions over D=32) ----
    for (int r = 0; r < 20; ++r) {
        const float xv = tok[r * DDIM + lane];
        float sum = xv;
        for (int off = 16; off > 0; off >>= 1) sum += __shfl_xor(sum, off, 32);
        const float mu = sum * (1.f / DDIM);
        const float dv = xv - mu;
        float s2 = dv * dv;
        for (int off = 16; off > 0; off >>= 1) s2 += __shfl_xor(s2, off, 32);
        const float var = s2 * (1.f / DDIM);
        tok[r * DDIM + lane] = dv * rsqrtf(var + 1e-5f) * ln1g[lane] + ln1b[lane];
    }
    __syncthreads();

    // ---- FFN: 32 -> 128 (relu) -> 32, hidden chunked in K=32 slices ----
    v8f acc[2][2] = {};
    {
        v16h aT0 = ldA_f32(tok, DDIM, 0, 0, lane);
        v16h aT1 = ldA_f32(tok, DDIM, 16, 0, lane);
#pragma unroll
        for (int cc = 0; cc < 4; ++cc) {
#pragma unroll
            for (int mt = 0; mt < 2; ++mt) {
                v16h a = (mt == 0) ? aT0 : aT1;
#pragma unroll
                for (int nt = 0; nt < 2; ++nt) {
                    v16h b = ldB_wT(s_w1, DDIM, cc * 32 + nt * 16, 0, lane);
                    v8f  c = {};
                    c = wmma16x16x32(a, b, c);
                    const float bb = b1[cc * 32 + nt * 16 + (lane & 15)];
#pragma unroll
                    for (int j = 0; j < 8; ++j) { const float t = c[j] + bb; c[j] = t > 0.f ? t : 0.f; }
                    stC_f16(shs, DDIM, mt * 16, nt * 16, lane, c);
                }
            }
            __syncthreads();
#pragma unroll
            for (int mt = 0; mt < 2; ++mt) {
                v16h a = ldA_f16(shs, DDIM, mt * 16, 0, lane);
#pragma unroll
                for (int nt = 0; nt < 2; ++nt) {
                    v16h b = ldB_wT(s_w2, DFFN, nt * 16, cc * 32, lane);
                    acc[mt][nt] = wmma16x16x32(a, b, acc[mt][nt]);
                }
            }
            __syncthreads();
        }
    }
    // residual + bias
#pragma unroll
    for (int mt = 0; mt < 2; ++mt) {
#pragma unroll
        for (int nt = 0; nt < 2; ++nt) {
            const int   n  = nt * 16 + (lane & 15);
            const int   mb = mt * 16 + ((lane >> 4) << 3);
            const float bb = b2[n];
#pragma unroll
            for (int j = 0; j < 8; ++j) tok[(mb + j) * DDIM + n] += acc[mt][nt][j] + bb;
        }
    }
    __syncthreads();

    // ---- LayerNorm 2 ----
    for (int r = 0; r < 20; ++r) {
        const float xv = tok[r * DDIM + lane];
        float sum = xv;
        for (int off = 16; off > 0; off >>= 1) sum += __shfl_xor(sum, off, 32);
        const float mu = sum * (1.f / DDIM);
        const float dv = xv - mu;
        float s2 = dv * dv;
        for (int off = 16; off > 0; off >>= 1) s2 += __shfl_xor(s2, off, 32);
        const float var = s2 * (1.f / DDIM);
        tok[r * DDIM + lane] = dv * rsqrtf(var + 1e-5f) * ln2g[lane] + ln2b[lane];
    }
    __syncthreads();

    // ---- store: channel tokens -> h_c_out, marker token -> hm ----
    for (int r = 0; r < CCH; ++r)
        hc_out[(size_t)s * CCH * DDIM + r * DDIM + lane] = tok[r * DDIM + lane];
    hm_out[(size_t)s * DDIM + lane] = tok[19 * DDIM + lane];
}

// =====================================================================
// Kernel 3: BiLSTM over T=120 + final linear.  One block per sequence
// (B*NSZ = 32 blocks).  Threads 0..63 = forward gates, 64..127 = backward.
// =====================================================================
__global__ __launch_bounds__(128) void k_lstm(
    const float* __restrict__ hm,
    const float* __restrict__ wih_f, const float* __restrict__ whh_f,
    const float* __restrict__ bih_f, const float* __restrict__ bhh_f,
    const float* __restrict__ wih_b, const float* __restrict__ whh_b,
    const float* __restrict__ bih_b, const float* __restrict__ bhh_b,
    const float* __restrict__ lin_w, const float* __restrict__ lin_b,
    float* __restrict__ proba)
{
    __shared__ float s_h[2][HIDN];
    __shared__ float s_c[2][HIDN];
    __shared__ float s_g[2][4 * HIDN];
    __shared__ float s_hall[2][TT][HIDN];

    const int tid = threadIdx.x;
    const int dir = tid >> 6;
    const int j   = tid & 63;
    const int bn  = blockIdx.x;                     // 0..31

    const float* wih = dir ? wih_b : wih_f;
    const float* whh = dir ? whh_b : whh_f;
    const float  bsum = dir ? (bih_b[j] + bhh_b[j]) : (bih_f[j] + bhh_f[j]);

    float wi[DDIM], wh[HIDN];
#pragma unroll
    for (int k = 0; k < DDIM; ++k) wi[k] = wih[j * DDIM + k];
#pragma unroll
    for (int k = 0; k < HIDN; ++k) wh[k] = whh[j * HIDN + k];

    if (j < HIDN) { s_h[dir][j] = 0.f; s_c[dir][j] = 0.f; }
    __syncthreads();

    const float* xb = hm + (size_t)bn * TT * DDIM;
    for (int step = 0; step < TT; ++step) {
        const int t = dir ? (TT - 1 - step) : step;
        const float* xt = xb + t * DDIM;
        float g = bsum;
#pragma unroll
        for (int k = 0; k < DDIM; ++k) g += wi[k] * xt[k];
#pragma unroll
        for (int k = 0; k < HIDN; ++k) g += wh[k] * s_h[dir][k];
        s_g[dir][j] = g;
        __syncthreads();
        if (j < HIDN) {
            const float ig = s_g[dir][j];
            const float fg = s_g[dir][j + HIDN];
            const float gg = s_g[dir][j + 2 * HIDN];
            const float og = s_g[dir][j + 3 * HIDN];
            const float cn = sigm(fg) * s_c[dir][j] + sigm(ig) * tanhf(gg);
            const float hn = sigm(og) * tanhf(cn);
            s_c[dir][j] = cn;
            s_h[dir][j] = hn;
            s_hall[dir][t][j] = hn;
        }
        __syncthreads();
    }

    // final linear: [hf, hb] (32) -> 2 logits per timestep
    for (int idx = tid; idx < TT * 2; idx += 128) {
        const int t = idx >> 1;
        const int o = idx & 1;
        float acc = lin_b[o];
#pragma unroll
        for (int k = 0; k < HIDN; ++k) acc += lin_w[o * 2 * HIDN + k] * s_hall[0][t][k];
#pragma unroll
        for (int k = 0; k < HIDN; ++k) acc += lin_w[o * 2 * HIDN + HIDN + k] * s_hall[1][t][k];
        proba[(size_t)bn * TT * 2 + t * 2 + o] = acc;
    }
}

// =====================================================================
extern "C" void kernel_launch(void* const* d_in, const int* in_sizes, int n_in,
                              void* d_out, int out_size, void* d_ws, size_t ws_size,
                              hipStream_t stream) {
    (void)in_sizes; (void)n_in; (void)out_size; (void)ws_size;

    const float* x     = (const float*)d_in[0];
    const float* pos   = (const float*)d_in[1];
    const float* c1w   = (const float*)d_in[2];
    const float* c1b   = (const float*)d_in[3];
    const float* c2w   = (const float*)d_in[4];
    const float* c2b   = (const float*)d_in[5];
    const float* wq    = (const float*)d_in[6];
    const float* bq    = (const float*)d_in[7];
    const float* wk    = (const float*)d_in[8];
    const float* bk    = (const float*)d_in[9];
    const float* wv    = (const float*)d_in[10];
    const float* bv    = (const float*)d_in[11];
    const float* wo    = (const float*)d_in[12];
    const float* bo    = (const float*)d_in[13];
    const float* ln1g  = (const float*)d_in[14];
    const float* ln1b  = (const float*)d_in[15];
    const float* w1    = (const float*)d_in[16];
    const float* b1    = (const float*)d_in[17];
    const float* w2    = (const float*)d_in[18];
    const float* b2    = (const float*)d_in[19];
    const float* ln2g  = (const float*)d_in[20];
    const float* ln2b  = (const float*)d_in[21];
    const float* wih_f = (const float*)d_in[22];
    const float* whh_f = (const float*)d_in[23];
    const float* bih_f = (const float*)d_in[24];
    const float* bhh_f = (const float*)d_in[25];
    const float* wih_b = (const float*)d_in[26];
    const float* whh_b = (const float*)d_in[27];
    const float* bih_b = (const float*)d_in[28];
    const float* bhh_b = (const float*)d_in[29];
    const float* lin_w = (const float*)d_in[30];
    const float* lin_b = (const float*)d_in[31];

    float* out    = (float*)d_out;
    float* proba  = out;                                   // B*NSZ*T*2 = 7680
    float* hc     = out + (size_t)BB * NSZv * TT * 2;      // Bt*19*32
    float* tokbuf = (float*)d_ws;                          // Bt*19*32 f32
    float* hmbuf  = tokbuf + (size_t)BT * CCH * DDIM;      // Bt*32 f32

    k_cnn<<<(BT * CCH) / 16, 256, 0, stream>>>(x, pos, c1w, c1b, c2w, c2b, tokbuf);
    k_xform<<<BT / 2, 64, 0, stream>>>(tokbuf, pos, wq, bq, wk, bk, wv, bv, wo, bo,
                                       ln1g, ln1b, w1, b1, w2, b2, ln2g, ln2b,
                                       hc, hmbuf);
    k_lstm<<<BB * NSZv, 128, 0, stream>>>(hmbuf, wih_f, whh_f, bih_f, bhh_f,
                                          wih_b, whh_b, bih_b, bhh_b,
                                          lin_w, lin_b, proba);
}
